// PointNetSetAbstraction_55456617726254
// MI455X (gfx1250) — matrix operand order repaired and checked by reading, not compiled, not run on hardware
//
#include <hip/hip_runtime.h>
#include <hip/hip_bf16.h>

// ---------------------------------------------------------------------------
// PointNet++ Set Abstraction for MI455X (gfx1250), wave32 + WMMA bf16.
// ---------------------------------------------------------------------------

typedef __attribute__((ext_vector_type(16))) __bf16 v16bf;
typedef __attribute__((ext_vector_type(8)))  float  v8f;

#define Bq 16
#define Np 4096
#define Sq 1024
#define Ksamp 32
#define Mrows (Bq * Sq * Ksamp)   // 524288 rows through the MLP

__device__ __forceinline__ unsigned short f2bf(float f) {
  unsigned int u = __builtin_bit_cast(unsigned int, f);
  u += 0x7FFFu + ((u >> 16) & 1u);            // round-to-nearest-even
  return (unsigned short)(u >> 16);
}
__device__ __forceinline__ float bf2f(unsigned short s) {
  unsigned int u = ((unsigned int)s) << 16;
  return __builtin_bit_cast(float, u);
}
__device__ __forceinline__ __bf16 us2bf(unsigned short s) {
  return __builtin_bit_cast(__bf16, s);
}

// ---------------------------------------------------------------------------
// Zero the BN statistics accumulators (3 layers x 512 floats).
// ---------------------------------------------------------------------------
__global__ void zero_stats_kernel(float* p) {
  int i = blockIdx.x * 256 + threadIdx.x;
  if (i < 3 * 512) p[i] = 0.0f;
}

// ---------------------------------------------------------------------------
// kNN ball query + group + center + concat -> x0 [M, 32] bf16 (cols 6..31 = 0)
// One wave per query point; 4 waves / block; 64 KB LDS of distances.
// Also writes new_xyz (first output) to d_out.
// ---------------------------------------------------------------------------
__global__ __launch_bounds__(128) void knn_group_kernel(
    const float* __restrict__ xyz, const float* __restrict__ pts,
    const int* __restrict__ fps, float* __restrict__ newxyz,
    unsigned short* __restrict__ x0) {
  __shared__ float dist[4 * Np];
  const int wave = threadIdx.x >> 5;
  const int lane = threadIdx.x & 31;
  const int q = blockIdx.x * 4 + wave;          // query id in [0, B*S)
  const int b = q >> 10;                        // / Sq
  const int s = q & (Sq - 1);

  const int qi = fps[b * Sq + s];
  const float* qp = xyz + ((size_t)b * Np + qi) * 3;
  const float qx = qp[0], qy = qp[1], qz = qp[2];
  if (lane < 3) newxyz[(size_t)q * 3 + lane] = (lane == 0) ? qx : ((lane == 1) ? qy : qz);

  const float* xb = xyz + (size_t)b * Np * 3;
  float* dw = dist + wave * Np;
  for (int j = 0; j < Np / 32; ++j) {
    const int n = lane + 32 * j;
    const float dx = xb[n * 3 + 0] - qx;
    const float dy = xb[n * 3 + 1] - qy;
    const float dz = xb[n * 3 + 2] - qz;
    dw[n] = dx * dx + dy * dy + dz * dz;
  }
  __syncthreads();

  // 32 selection passes: pick min (d, idx) strictly after the last selected.
  float lastd = -1.0f;
  int lasti = -1;
  int myidx = 0;
  for (int p = 0; p < Ksamp; ++p) {
    float bd = 3.0e38f;
    int bi = 0x7FFFFFFF;
    for (int j = 0; j < Np / 32; ++j) {
      const int n = lane + 32 * j;
      const float d = dw[n];
      const bool after = (d > lastd) || (d == lastd && n > lasti);
      const bool better = (d < bd) || (d == bd && n < bi);
      if (after && better) { bd = d; bi = n; }
    }
#pragma unroll
    for (int off = 16; off > 0; off >>= 1) {
      const float od = __shfl_xor(bd, off, 32);
      const int   oi = __shfl_xor(bi, off, 32);
      if (od < bd || (od == bd && oi < bi)) { bd = od; bi = oi; }
    }
    if (lane == p) myidx = bi;
    lastd = bd;
    lasti = bi;
  }

  // Lane k gathers neighbor k, centers xyz, concatenates point features.
  const float* pp = xb + (size_t)myidx * 3;
  const float* fp = pts + ((size_t)b * Np + myidx) * 3;
  const float v0 = pp[0] - qx, v1 = pp[1] - qy, v2 = pp[2] - qz;
  const float v3 = fp[0], v4 = fp[1], v5 = fp[2];

  unsigned int* ou = (unsigned int*)(x0 + ((size_t)q * Ksamp + lane) * 32);
  ou[0] = (unsigned int)f2bf(v0) | ((unsigned int)f2bf(v1) << 16);
  ou[1] = (unsigned int)f2bf(v2) | ((unsigned int)f2bf(v3) << 16);
  ou[2] = (unsigned int)f2bf(v4) | ((unsigned int)f2bf(v5) << 16);
#pragma unroll
  for (int i = 3; i < 16; ++i) ou[i] = 0u;      // zero pad channels 6..31
}

// ---------------------------------------------------------------------------
// GEMM (bf16 WMMA, f32 acc): Y[M,Cout] = A[M,Cin] * W^T + bias, Y stored bf16.
// Fused per-channel sum / sum-of-squares for BatchNorm batch statistics.
// Block = 256 threads = 8 waves; each block owns 128 rows; each wave owns a
// 16-row subtile. A fragments (depend only on K-chunk) are hoisted into
// registers once per wave; the column-tile loop loads only B fragments, so
// the steady state is 2x ds_load_b128 + 1x v_wmma per K-chunk.
// ---------------------------------------------------------------------------
template <int CIN, int CINW, int COUT>
__global__ __launch_bounds__(256) void gemm_bn_kernel(
    const unsigned short* __restrict__ A,
    const float* __restrict__ W,
    const float* __restrict__ bias,
    unsigned short* __restrict__ Y,
    float* __restrict__ stats) {                     // [0]=sum [128]=sumsq
  __shared__ unsigned short a_sh[128 * CIN];
  __shared__ unsigned short w_sh[COUT * CIN];
  __shared__ float s_sum[128];
  __shared__ float s_sq[128];

  const int tid = threadIdx.x;
  if (tid < 128) { s_sum[tid] = 0.0f; s_sq[tid] = 0.0f; }

  const size_t m0 = (size_t)blockIdx.x * 128;

  // Stage A tile (rows are contiguous bf16) into LDS.
  {
    const unsigned int* src = (const unsigned int*)(A + m0 * CIN);
    unsigned int* dst = (unsigned int*)a_sh;
    constexpr int total = 128 * CIN / 2;
#pragma unroll
    for (int i = 0; i < total / 256; ++i) dst[tid + 256 * i] = src[tid + 256 * i];
    if (tid == 0 && (int)blockIdx.x + 1 < (int)gridDim.x)
      __builtin_prefetch((const char*)(A + (m0 + 128) * CIN), 0, 0);
  }
  // Stage weights, converted f32 -> bf16, zero-padded K (layer 0: 6 -> 32).
  {
    constexpr int total = COUT * CIN;
    for (int i = tid; i < total; i += 256) {
      const int n = i / CIN;
      const int k = i - n * CIN;
      w_sh[i] = (k < CINW) ? f2bf(W[n * CINW + k]) : (unsigned short)0;
    }
  }
  __syncthreads();

  const int wv = tid >> 5;
  const int lane = tid & 31;
  const int m_sub = wv * 16;
  const int g = lane >> 4;       // lane half
  const int l15 = lane & 15;
  constexpr int NKC = CIN / 32;
  constexpr int NT = COUT / 16;

  // Hoist A fragments: this wave's 16 rows x CIN, kept in VGPRs.
  // (contiguous k order, same as B): row = l15, K = 16g + (0..15).
  v16bf afrag[NKC];
#pragma unroll
  for (int kc = 0; kc < NKC; ++kc) {
    const unsigned short* pa = a_sh + (m_sub + l15) * CIN + kc * 32 + 16 * g;
#pragma unroll
    for (int i = 0; i < 8; ++i) afrag[kc][i] = us2bf(pa[i]);
#pragma unroll
    for (int i = 0; i < 8; ++i) afrag[kc][8 + i] = us2bf(pa[8 + i]);
  }

#pragma unroll
  for (int ct = 0; ct < NT; ++ct) {
    v8f acc = {};
#pragma unroll
    for (int kc = 0; kc < NKC; ++kc) {
      // B fragment: B[k][n] = W[n][k]; n = l15, K = 16g + (0..15).
      v16bf bf;
      const unsigned short* pb = w_sh + (ct * 16 + l15) * CIN + kc * 32 + 16 * g;
#pragma unroll
      for (int i = 0; i < 16; ++i) bf[i] = us2bf(pb[i]);
      acc = __builtin_amdgcn_wmma_f32_16x16x32_bf16(
          false, afrag[kc], false, bf, (short)0, acc, false, false);
    }
    // C/D layout: element e -> m = e + 8g, n = l15.
    const float bn = bias[ct * 16 + l15];
    float s1 = 0.0f, s2 = 0.0f;
#pragma unroll
    for (int e = 0; e < 8; ++e) {
      const float y = acc[e] + bn;
      s1 += y;
      s2 += y * y;
      const size_t row = m0 + m_sub + e + 8 * g;
      Y[row * COUT + ct * 16 + l15] = f2bf(y);
    }
    s1 += __shfl_xor(s1, 16, 32);   // lanes L and L+16 share channel n
    s2 += __shfl_xor(s2, 16, 32);
    if (g == 0) {
      atomicAdd(&s_sum[ct * 16 + l15], s1);
      atomicAdd(&s_sq[ct * 16 + l15], s2);
    }
  }
  __syncthreads();
  for (int c = tid; c < COUT; c += 256) {
    atomicAdd(&stats[c], s_sum[c]);
    atomicAdd(&stats[128 + c], s_sq[c]);
  }
}

// ---------------------------------------------------------------------------
// BN finalize: per-channel scale/shift from accumulated sums.
// ---------------------------------------------------------------------------
__global__ void finalize_kernel(float* stats, const float* __restrict__ gw,
                                const float* __restrict__ tw, int Cout,
                                float invM) {
  const int c = threadIdx.x;
  if (c < Cout) {
    const float mean = stats[c] * invM;
    const float var = stats[128 + c] * invM - mean * mean;
    const float sc = gw[c] * rsqrtf(var + 1e-5f);
    stats[256 + c] = sc;
    stats[384 + c] = tw[c] - mean * sc;
  }
}

// ---------------------------------------------------------------------------
// In-place normalize + ReLU on bf16 activations.
// ---------------------------------------------------------------------------
__global__ __launch_bounds__(256) void norm_relu_kernel(
    unsigned short* __restrict__ Y, const float* __restrict__ stats,
    int cmask) {
  const size_t i = (size_t)blockIdx.x * 256 + threadIdx.x;
  const int c = (int)(i & (size_t)cmask);
  float y = bf2f(Y[i]);
  y = y * stats[256 + c] + stats[384 + c];
  Y[i] = f2bf(fmaxf(y, 0.0f));
}

// ---------------------------------------------------------------------------
// Final layer: normalize + ReLU + max over K=32 samples -> feat f32 to d_out.
// One block per (b,s); 128 threads = 128 channels; coalesced per-k reads.
// ---------------------------------------------------------------------------
__global__ __launch_bounds__(128) void norm_maxpool_kernel(
    const unsigned short* __restrict__ Y, const float* __restrict__ stats,
    float* __restrict__ out) {
  const int bs = blockIdx.x;
  const int c = threadIdx.x;
  const float sc = stats[256 + c];
  const float sh = stats[384 + c];
  const unsigned short* p = Y + (size_t)bs * Ksamp * 128 + c;
  float m = -3.0e38f;
#pragma unroll
  for (int k = 0; k < Ksamp; ++k) {
    m = fmaxf(m, bf2f(p[k * 128]) * sc + sh);
  }
  out[(size_t)bs * 128 + c] = fmaxf(m, 0.0f);   // ReLU commutes with max
}

// ---------------------------------------------------------------------------
// Host-side launch sequence (graph-capture safe, deterministic).
// ---------------------------------------------------------------------------
extern "C" void kernel_launch(void* const* d_in, const int* in_sizes, int n_in,
                              void* d_out, int out_size, void* d_ws,
                              size_t ws_size, hipStream_t stream) {
  (void)in_sizes; (void)n_in; (void)out_size; (void)ws_size;
  const float* xyz = (const float*)d_in[0];
  const float* pts = (const float*)d_in[1];
  const int* fps   = (const int*)d_in[2];
  const float* w0 = (const float*)d_in[3],  *b0 = (const float*)d_in[4];
  const float* g0 = (const float*)d_in[5],  *t0 = (const float*)d_in[6];
  const float* w1 = (const float*)d_in[7],  *b1 = (const float*)d_in[8];
  const float* g1 = (const float*)d_in[9],  *t1 = (const float*)d_in[10];
  const float* w2 = (const float*)d_in[11], *b2 = (const float*)d_in[12];
  const float* g2 = (const float*)d_in[13], *t2 = (const float*)d_in[14];

  float* outf = (float*)d_out;
  char* ws = (char*)d_ws;
  float* stats0 = (float*)ws;          // 512 floats per layer
  float* stats1 = stats0 + 512;
  float* stats2 = stats1 + 512;
  unsigned short* bufA = (unsigned short*)(ws + 8192);                          // M*64 bf16
  unsigned short* bufB = (unsigned short*)(ws + 8192 + (size_t)Mrows * 64 * 2); // M*128 bf16
  const float invM = 1.0f / (float)Mrows;

  zero_stats_kernel<<<6, 256, 0, stream>>>((float*)ws);

  // kNN + grouping -> x0 in bufA (stride 32), new_xyz -> d_out[0:49152)
  knn_group_kernel<<<(Bq * Sq) / 4, 128, 0, stream>>>(xyz, pts, fps, outf, bufA);

  // Layer 0: [M,32(pad of 6)] x [6,64] -> bufB
  gemm_bn_kernel<32, 6, 64><<<Mrows / 128, 256, 0, stream>>>(bufA, w0, b0, bufB, stats0);
  finalize_kernel<<<1, 128, 0, stream>>>(stats0, g0, t0, 64, invM);
  norm_relu_kernel<<<(Mrows * 64) / 256, 256, 0, stream>>>(bufB, stats0, 63);

  // Layer 1: [M,64] x [64,64] -> bufA
  gemm_bn_kernel<64, 64, 64><<<Mrows / 128, 256, 0, stream>>>(bufB, w1, b1, bufA, stats1);
  finalize_kernel<<<1, 128, 0, stream>>>(stats1, g1, t1, 64, invM);
  norm_relu_kernel<<<(Mrows * 64) / 256, 256, 0, stream>>>(bufA, stats1, 63);

  // Layer 2: [M,64] x [64,128] -> bufB
  gemm_bn_kernel<64, 64, 128><<<Mrows / 128, 256, 0, stream>>>(bufA, w2, b2, bufB, stats2);
  finalize_kernel<<<1, 128, 0, stream>>>(stats2, g2, t2, 128, invM);

  // Normalize + ReLU + max over K -> feat f32 at d_out[49152:]
  norm_maxpool_kernel<<<Bq * Sq, 128, 0, stream>>>(bufB, stats2, outf + Bq * Sq * 3);
}
